// SecondAttn_16793322128107
// MI455X (gfx1250) — matrix operand order repaired and checked by reading, not compiled, not run on hardware
//
#include <hip/hip_runtime.h>
#include <hip/hip_bf16.h>

// ---------------------------------------------------------------------------
// Cross-attention (B=1, N=M=4096, DIM=512, HEADS=8, HEAD_DIM=64)
// Outputs (concatenated in d_out): out [4096*512] fp32, attn [8*4096*4096] fp32
// Bandwidth-bound on the materialized 537MB attn tensor (>= ~2.1GB HBM traffic
// ~= 90us at 23.3TB/s). All GEMMs on v_wmma_f32_16x16x32_f16; f16 tile copies
// use gfx1250 GLOBAL_LOAD_ASYNC_TO_LDS (ASYNCcnt) so the VALU only does the
// unavoidable fp32<->f16 conversion work on the attn stream.
// ---------------------------------------------------------------------------

#define NQ 4096
#define NM 4096
#define DIM 512
#define HEADS 8
#define HD 64
#define ATTN_SCALE 0.125f  // 64^-0.5

typedef __attribute__((ext_vector_type(16))) _Float16 v16h;
typedef __attribute__((ext_vector_type(8)))  float    v8f;

union H16 { v16h v; _Float16 h[16]; };
union F8  { v8f  v; float    f[8];  };

// --- gfx1250 async global->LDS copy (16B per lane), tracked on ASYNCcnt -----
__device__ __forceinline__ void async_copy_b128(void* lds_dst, const void* gsrc) {
    const unsigned lds           = (unsigned)(unsigned long long)lds_dst;
    const unsigned long long gaddr = (unsigned long long)gsrc;
    asm volatile("global_load_async_to_lds_b128 %0, %1, off"
                 :: "v"(lds), "v"(gaddr) : "memory");
}
__device__ __forceinline__ void wait_asynccnt0() {
    asm volatile("s_wait_asynccnt 0x0" ::: "memory");
}

// --- WMMA fragment loaders (contiguous k order, same for A and B, wave32) -----------

// A matrix 16x32 f16: lane m=lane&15 holds row m; VGPR v in [0,8):
//   k = (lane>>4)*16 + 2v  (pair k, k+1)
__device__ __forceinline__ v16h load_A_lds(const _Float16* base, int ld, int koff, int lane) {
    H16 a;
    const int m = lane & 15, hi = lane >> 4;
    #pragma unroll
    for (int v = 0; v < 8; ++v) {
        const int k = koff + hi * 16 + 2 * v;
        a.h[2 * v]     = base[m * ld + k];
        a.h[2 * v + 1] = base[m * ld + k + 1];
    }
    return a.v;
}

// B matrix 32x16 f16, source stored as [col][k] (i.e. we need B = srcT):
//   lane n=lane&15 holds column n; VGPR v: k = (lane>>4)*16 + 2v  (pair)
__device__ __forceinline__ v16h load_B_ldsT(const _Float16* base, int ld, int koff, int lane) {
    H16 b;
    const int n = lane & 15, hi = lane >> 4;
    #pragma unroll
    for (int v = 0; v < 8; ++v) {
        const int k = koff + hi * 16 + 2 * v;
        b.h[2 * v]     = base[n * ld + k];
        b.h[2 * v + 1] = base[n * ld + k + 1];
    }
    return b.v;
}

// B matrix 32x16 f16, source stored row-major [k][n]
__device__ __forceinline__ v16h load_B_lds_rm(const _Float16* base, int ld, int lane) {
    H16 b;
    const int n = lane & 15, hi = lane >> 4;
    #pragma unroll
    for (int v = 0; v < 8; ++v) {
        const int k = hi * 16 + 2 * v;
        b.h[2 * v]     = base[k * ld + n];
        b.h[2 * v + 1] = base[(k + 1) * ld + n];
    }
    return b.v;
}

// ---------------------------------------------------------------------------
// Kernel 1: fused embed + QKV projection.  Y = (X + emb) @ W^T, f16 out.
// grid (128, 8, 3): 32-row x 64-col tile; z: 0->Q, 1->K, 2->V
// (fp32 source needs VALU conversion anyway -> classic staged path)
// ---------------------------------------------------------------------------
__global__ __launch_bounds__(256)
void qkv_proj_kernel(const float* __restrict__ x, const float* __restrict__ ctx,
                     const float* __restrict__ Wq, const float* __restrict__ Wkv,
                     const float* __restrict__ emb,
                     _Float16* __restrict__ Q, _Float16* __restrict__ K,
                     _Float16* __restrict__ V) {
    const int job = blockIdx.z;
    const float* X = (job == 0) ? x : ctx;
    const float* W = (job == 0) ? Wq : (job == 1 ? Wkv : Wkv + DIM * DIM);
    _Float16* Y = (job == 0) ? Q : (job == 1 ? K : V);

    const int row0 = blockIdx.x * 32;
    const int col0 = blockIdx.y * 64;

    __shared__ _Float16 Xs[32][40];
    __shared__ _Float16 Ws[64][40];

    const int tid = threadIdx.x;
    const int lane = tid & 31, wave = tid >> 5;
    const int wr = wave >> 2, wc = wave & 3;   // 2x4 waves -> 32x64 tile

    F8 c;
    #pragma unroll
    for (int i = 0; i < 8; ++i) c.f[i] = 0.f;

    for (int k0 = 0; k0 < DIM; k0 += 32) {
        {   // stage X tile 32x32 fp32 -> f16 (+emb)
            const int r = tid >> 3, cc = (tid & 7) * 4;
            const float4 xv = *(const float4*)(X + (size_t)(row0 + r) * DIM + k0 + cc);
            Xs[r][cc + 0] = (_Float16)(xv.x + emb[k0 + cc + 0]);
            Xs[r][cc + 1] = (_Float16)(xv.y + emb[k0 + cc + 1]);
            Xs[r][cc + 2] = (_Float16)(xv.z + emb[k0 + cc + 2]);
            Xs[r][cc + 3] = (_Float16)(xv.w + emb[k0 + cc + 3]);
        }
        {   // stage W tile 64x32 fp32 -> f16  (rows = output cols)
            const int r = tid >> 2, cc = (tid & 3) * 8;
            const float4 w0 = *(const float4*)(W + (size_t)(col0 + r) * DIM + k0 + cc);
            const float4 w1 = *(const float4*)(W + (size_t)(col0 + r) * DIM + k0 + cc + 4);
            Ws[r][cc + 0] = (_Float16)w0.x; Ws[r][cc + 1] = (_Float16)w0.y;
            Ws[r][cc + 2] = (_Float16)w0.z; Ws[r][cc + 3] = (_Float16)w0.w;
            Ws[r][cc + 4] = (_Float16)w1.x; Ws[r][cc + 5] = (_Float16)w1.y;
            Ws[r][cc + 6] = (_Float16)w1.z; Ws[r][cc + 7] = (_Float16)w1.w;
        }
        __syncthreads();

        const v16h a = load_A_lds(&Xs[wr * 16][0], 40, 0, lane);
        const v16h b = load_B_ldsT(&Ws[wc * 16][0], 40, 0, lane);
        c.v = __builtin_amdgcn_wmma_f32_16x16x32_f16(false, a, false, b,
                                                     (short)0, c.v, false, false);
        __syncthreads();
    }

    const int n = lane & 15, hi = lane >> 4;
    #pragma unroll
    for (int r = 0; r < 8; ++r) {
        Y[(size_t)(row0 + wr * 16 + r + 8 * hi) * DIM + col0 + wc * 16 + n] =
            (_Float16)c.f[r];
    }
}

// ---------------------------------------------------------------------------
// Kernel 2: S = Q_h @ K_h^T * scale  (raw scores -> attn region, fp32)
// grid (64, 64, HEADS): 64x64 score tile per block.  Q/K tiles arrive via
// async global->LDS DMA; each wave drives 2 accumulators (B-frag reuse).
// ---------------------------------------------------------------------------
__global__ __launch_bounds__(256)
void scores_kernel(const _Float16* __restrict__ Q, const _Float16* __restrict__ K,
                   float* __restrict__ attn) {
    const int h = blockIdx.z;
    const int row0 = blockIdx.x * 64;
    const int col0 = blockIdx.y * 64;

    __shared__ _Float16 Qs[64][80];
    __shared__ _Float16 Ks[64][80];

    const int tid = threadIdx.x;
    const int lane = tid & 31, wave = tid >> 5;
    const int wr = wave >> 2, wc = wave & 3;

    {   // async stage Q tile 64x64 f16 and K tile 64x64 f16 (16B/lane x4)
        const int r = tid >> 2, cc = (tid & 3) * 16;
        async_copy_b128(&Qs[r][cc],     Q + (size_t)(row0 + r) * DIM + h * HD + cc);
        async_copy_b128(&Qs[r][cc + 8], Q + (size_t)(row0 + r) * DIM + h * HD + cc + 8);
        async_copy_b128(&Ks[r][cc],     K + (size_t)(col0 + r) * DIM + h * HD + cc);
        async_copy_b128(&Ks[r][cc + 8], K + (size_t)(col0 + r) * DIM + h * HD + cc + 8);
    }
    wait_asynccnt0();
    __syncthreads();

    F8 c0, c1;
    #pragma unroll
    for (int i = 0; i < 8; ++i) { c0.f[i] = 0.f; c1.f[i] = 0.f; }

    #pragma unroll
    for (int ks = 0; ks < HD; ks += 32) {
        const v16h a0 = load_A_lds(&Qs[wr * 16][0], 80, ks, lane);
        const v16h a1 = load_A_lds(&Qs[32 + wr * 16][0], 80, ks, lane);
        const v16h b  = load_B_ldsT(&Ks[wc * 16][0], 80, ks, lane);
        c0.v = __builtin_amdgcn_wmma_f32_16x16x32_f16(false, a0, false, b,
                                                      (short)0, c0.v, false, false);
        c1.v = __builtin_amdgcn_wmma_f32_16x16x32_f16(false, a1, false, b,
                                                      (short)0, c1.v, false, false);
    }

    float* S = attn + (size_t)h * NQ * NM;
    const int n = lane & 15, hi = lane >> 4;
    #pragma unroll
    for (int r = 0; r < 8; ++r) {
        const size_t col = col0 + wc * 16 + n;
        S[(size_t)(row0 + wr * 16 + r + 8 * hi) * NM + col]      = c0.f[r] * ATTN_SCALE;
        S[(size_t)(row0 + 32 + wr * 16 + r + 8 * hi) * NM + col] = c1.f[r] * ATTN_SCALE;
    }
}

// ---------------------------------------------------------------------------
// Kernel 3: row softmax in place over attn (8*4096 rows of 4096).
// One block per row; 16 values/thread live in registers -> 1 read + 1 write.
// ---------------------------------------------------------------------------
__global__ __launch_bounds__(256)
void softmax_kernel(float* __restrict__ attn) {
    float* p = attn + (size_t)blockIdx.x * NM;
    const int tid = threadIdx.x;

    float vals[16];
    float mx = -1e30f;
    #pragma unroll
    for (int i = 0; i < 16; ++i) {
        vals[i] = p[tid + i * 256];
        mx = fmaxf(mx, vals[i]);
    }
    #pragma unroll
    for (int off = 16; off > 0; off >>= 1)
        mx = fmaxf(mx, __shfl_xor(mx, off, 32));

    __shared__ float red_max[8];
    __shared__ float red_sum[8];
    if ((tid & 31) == 0) red_max[tid >> 5] = mx;
    __syncthreads();
    float rm = red_max[0];
    #pragma unroll
    for (int i = 1; i < 8; ++i) rm = fmaxf(rm, red_max[i]);

    float sum = 0.f;
    #pragma unroll
    for (int i = 0; i < 16; ++i) {
        vals[i] = __expf(vals[i] - rm);
        sum += vals[i];
    }
    #pragma unroll
    for (int off = 16; off > 0; off >>= 1)
        sum += __shfl_xor(sum, off, 32);
    if ((tid & 31) == 0) red_sum[tid >> 5] = sum;
    __syncthreads();
    float rs = 0.f;
    #pragma unroll
    for (int i = 0; i < 8; ++i) rs += red_sum[i];

    const float inv = 1.0f / rs;
    #pragma unroll
    for (int i = 0; i < 16; ++i) p[tid + i * 256] = vals[i] * inv;
}

// ---------------------------------------------------------------------------
// Kernel 4: out_pre = P @ V per head.  grid (64, HEADS): 64-row x 64-col tile.
// V tiles via async global->LDS DMA, overlapped with the VALU fp32->f16
// conversion of the P tile (the big 537MB re-read).  2 accumulators/wave.
// ---------------------------------------------------------------------------
__global__ __launch_bounds__(256)
void pv_kernel(const float* __restrict__ attn, const _Float16* __restrict__ V,
               float* __restrict__ outp) {
    const int h = blockIdx.y;
    const int row0 = blockIdx.x * 64;
    const float* P = attn + (size_t)h * NQ * NM;

    __shared__ _Float16 Ps[64][40];
    __shared__ _Float16 Vs[32][80];

    const int tid = threadIdx.x;
    const int lane = tid & 31, wave = tid >> 5;
    const int wr = wave >> 2, wc = wave & 3;

    F8 c0, c1;
    #pragma unroll
    for (int i = 0; i < 8; ++i) { c0.f[i] = 0.f; c1.f[i] = 0.f; }

    for (int m0 = 0; m0 < NM; m0 += 32) {
        {   // async stage V tile 32x64 f16 (no VGPR round-trip)
            const int r = tid >> 3, cc = (tid & 7) * 8;
            async_copy_b128(&Vs[r][cc], V + (size_t)(m0 + r) * DIM + h * HD + cc);
        }
        {   // stage P tile 64x32 fp32 -> f16 on the VALU (overlaps async V)
            const int r = tid >> 2, cc = (tid & 3) * 8;
            const float* src = P + (size_t)(row0 + r) * NM + m0 + cc;
            const float4 p0 = *(const float4*)src;
            const float4 p1 = *(const float4*)(src + 4);
            __builtin_prefetch(src + 32, 0, 1);   // next tile of the stream
            Ps[r][cc + 0] = (_Float16)p0.x; Ps[r][cc + 1] = (_Float16)p0.y;
            Ps[r][cc + 2] = (_Float16)p0.z; Ps[r][cc + 3] = (_Float16)p0.w;
            Ps[r][cc + 4] = (_Float16)p1.x; Ps[r][cc + 5] = (_Float16)p1.y;
            Ps[r][cc + 6] = (_Float16)p1.z; Ps[r][cc + 7] = (_Float16)p1.w;
        }
        wait_asynccnt0();
        __syncthreads();

        const v16h a0 = load_A_lds(&Ps[wr * 16][0], 40, 0, lane);
        const v16h a1 = load_A_lds(&Ps[32 + wr * 16][0], 40, 0, lane);
        const v16h b  = load_B_lds_rm(&Vs[0][wc * 16], 80, lane);
        c0.v = __builtin_amdgcn_wmma_f32_16x16x32_f16(false, a0, false, b,
                                                      (short)0, c0.v, false, false);
        c1.v = __builtin_amdgcn_wmma_f32_16x16x32_f16(false, a1, false, b,
                                                      (short)0, c1.v, false, false);
        __syncthreads();
    }

    const int n = lane & 15, hi = lane >> 4;
    #pragma unroll
    for (int r = 0; r < 8; ++r) {
        const size_t col = h * HD + wc * 16 + n;
        outp[(size_t)(row0 + wr * 16 + r + 8 * hi) * DIM + col]      = c0.f[r];
        outp[(size_t)(row0 + 32 + wr * 16 + r + 8 * hi) * DIM + col] = c1.f[r];
    }
}

// ---------------------------------------------------------------------------
// Kernel 5: out = out_pre @ Wproj^T + bproj  (fp32 out).  grid (128, 8).
// ---------------------------------------------------------------------------
__global__ __launch_bounds__(256)
void proj_kernel(const float* __restrict__ Xp, const float* __restrict__ Wp,
                 const float* __restrict__ bias, float* __restrict__ out) {
    const int row0 = blockIdx.x * 32;
    const int col0 = blockIdx.y * 64;

    __shared__ _Float16 Xs[32][40];
    __shared__ _Float16 Ws[64][40];

    const int tid = threadIdx.x;
    const int lane = tid & 31, wave = tid >> 5;
    const int wr = wave >> 2, wc = wave & 3;

    F8 c;
    #pragma unroll
    for (int i = 0; i < 8; ++i) c.f[i] = 0.f;

    for (int k0 = 0; k0 < DIM; k0 += 32) {
        {
            const int r = tid >> 3, cc = (tid & 7) * 4;
            const float4 xv = *(const float4*)(Xp + (size_t)(row0 + r) * DIM + k0 + cc);
            Xs[r][cc + 0] = (_Float16)xv.x; Xs[r][cc + 1] = (_Float16)xv.y;
            Xs[r][cc + 2] = (_Float16)xv.z; Xs[r][cc + 3] = (_Float16)xv.w;
        }
        {
            const int r = tid >> 2, cc = (tid & 3) * 8;
            const float4 w0 = *(const float4*)(Wp + (size_t)(col0 + r) * DIM + k0 + cc);
            const float4 w1 = *(const float4*)(Wp + (size_t)(col0 + r) * DIM + k0 + cc + 4);
            Ws[r][cc + 0] = (_Float16)w0.x; Ws[r][cc + 1] = (_Float16)w0.y;
            Ws[r][cc + 2] = (_Float16)w0.z; Ws[r][cc + 3] = (_Float16)w0.w;
            Ws[r][cc + 4] = (_Float16)w1.x; Ws[r][cc + 5] = (_Float16)w1.y;
            Ws[r][cc + 6] = (_Float16)w1.z; Ws[r][cc + 7] = (_Float16)w1.w;
        }
        __syncthreads();

        const v16h a = load_A_lds(&Xs[wr * 16][0], 40, 0, lane);
        const v16h b = load_B_ldsT(&Ws[wc * 16][0], 40, 0, lane);
        c.v = __builtin_amdgcn_wmma_f32_16x16x32_f16(false, a, false, b,
                                                     (short)0, c.v, false, false);
        __syncthreads();
    }

    const int n = lane & 15, hi = lane >> 4;
    #pragma unroll
    for (int r = 0; r < 8; ++r) {
        const int col = col0 + wc * 16 + n;
        out[(size_t)(row0 + wr * 16 + r + 8 * hi) * DIM + col] = c.f[r] + bias[col];
    }
}

// ---------------------------------------------------------------------------

extern "C" void kernel_launch(void* const* d_in, const int* in_sizes, int n_in,
                              void* d_out, int out_size, void* d_ws, size_t ws_size,
                              hipStream_t stream) {
    const float* x     = (const float*)d_in[0];
    const float* ctx   = (const float*)d_in[1];
    const float* Wq    = (const float*)d_in[2];
    const float* Wkv   = (const float*)d_in[3];
    const float* Wproj = (const float*)d_in[4];
    const float* bproj = (const float*)d_in[5];
    const float* emb   = (const float*)d_in[6];

    float* out  = (float*)d_out;                       // 4096*512 fp32
    float* attn = out + (size_t)NQ * DIM;              // 8*4096*4096 fp32

    // workspace: Q/K/V in f16 (4MB each) + out_pre fp32 (8MB) = 20MB
    _Float16* Qf = (_Float16*)d_ws;
    _Float16* Kf = Qf + (size_t)NQ * DIM;
    _Float16* Vf = Kf + (size_t)NM * DIM;
    float*  outp = (float*)(Vf + (size_t)NM * DIM);

    qkv_proj_kernel<<<dim3(NQ / 32, DIM / 64, 3), 256, 0, stream>>>(
        x, ctx, Wq, Wkv, emb, Qf, Kf, Vf);

    scores_kernel<<<dim3(NQ / 64, NM / 64, HEADS), 256, 0, stream>>>(Qf, Kf, attn);

    softmax_kernel<<<dim3(HEADS * NQ), 256, 0, stream>>>(attn);

    pv_kernel<<<dim3(NQ / 64, HEADS), 256, 0, stream>>>(attn, Vf, outp);

    proj_kernel<<<dim3(NQ / 32, DIM / 64), 256, 0, stream>>>(outp, Wproj, bproj, out);
}
